// PentaLinearFast_16716012716243
// MI455X (gfx1250) — matrix operand order repaired and checked by reading, not compiled, not run on hardware
//
#include <hip/hip_runtime.h>
#include <hip/hip_bf16.h>
#include <stdint.h>

// ---------------------------------------------------------------------------
// Penta-quantized linear: y = x @ dequant(packed_w).T + bias
// Kernel 1: unpack 3-bit codes -> bf16 weights in d_ws (once; ~6.7MB read)
// Kernel 2: bf16 WMMA GEMM, double-buffered LDS fed by CDNA5 async-to-LDS
//           copies (ASYNCcnt), overlapping HBM/L2 traffic with v_wmma work.
// ---------------------------------------------------------------------------

typedef __attribute__((ext_vector_type(16))) __bf16 v16bf;
typedef __attribute__((ext_vector_type(8)))  float  v8f;

union FragU { uint4 u[2]; v16bf v; };

#define BM 128
#define BN 128
#define BK 64
#define LDSP 72   // padded LDS row stride (bf16): 144B = 36 dwords, conflict-free

// CDNA5 async global->LDS copy, 16B per lane, tracked by ASYNCcnt.
__device__ __forceinline__ void async_copy_b128(unsigned lds_off, const void* gptr) {
    asm volatile("global_load_async_to_lds_b128 %0, %1, off"
                 :: "v"(lds_off), "v"(gptr) : "memory");
}
__device__ __forceinline__ void wait_async0() {
    asm volatile("s_wait_asynccnt 0x0" ::: "memory");
}
__device__ __forceinline__ unsigned lds_addr(const void* p) {
    // generic (shared) pointer low 32 bits == LDS byte offset (ISA aperture rule)
    return (unsigned)(uintptr_t)p;
}

// -------------------------- dequant pre-pass -------------------------------
__global__ __launch_bounds__(256) void penta_dequant_kernel(
    const int* __restrict__ pw, const float* __restrict__ scale,
    __bf16* __restrict__ wd, int K, int K_packs, int total_words)
{
    int widx = blockIdx.x * 256 + threadIdx.x;
    if (widx >= total_words) return;
    int n = widx / K_packs;
    int w = widx - n * K_packs;
    unsigned word = (unsigned)pw[widx];
    float s = scale[0];
    int kbase = w * 10;                       // even -> dword-aligned output
    unsigned* out32 = (unsigned*)(wd + (size_t)n * K + kbase);
#pragma unroll
    for (int j = 0; j < 10; j += 2) {
        if (kbase + j + 1 < K) {              // K even; pairs never straddle end
            float f0 = (float)((int)((word >> (3 * j))     & 7u) - 2) * s;
            float f1 = (float)((int)((word >> (3 * j + 3)) & 7u) - 2) * s;
            __bf16 h0 = (__bf16)f0, h1 = (__bf16)f1;
            unsigned b0 = __builtin_bit_cast(unsigned short, h0);
            unsigned b1 = __builtin_bit_cast(unsigned short, h1);
            out32[j >> 1] = b0 | (b1 << 16);
        }
    }
}

// ----------------------------- WMMA GEMM -----------------------------------
// grid: (N/BN, tokens/BM), block: 256 threads = 8 wave32 (2 M-waves x 4 N-waves)
__global__ __launch_bounds__(256) void penta_gemm_kernel(
    const __bf16* __restrict__ X,   // [tokens, K] row-major
    const __bf16* __restrict__ W,   // [N, K] row-major (dequantized, L2-hot)
    const float*  __restrict__ bias,
    __bf16* __restrict__ Y,         // [tokens, N]
    int Nf, int K)
{
    __shared__ __align__(16) __bf16 Xs[2][BM * LDSP];
    __shared__ __align__(16) __bf16 Ws[2][BN * LDSP];

    const int tid  = threadIdx.x;
    const int lane = tid & 31;
    const int wave = tid >> 5;
    const int bm = blockIdx.y * BM;
    const int bn = blockIdx.x * BN;
    const int wm = wave >> 2;   // 0..1 : 64-row M strip
    const int wn = wave & 3;    // 0..3 : 32-col N strip

    v8f acc[4][2];
#pragma unroll
    for (int i = 0; i < 4; ++i)
#pragma unroll
        for (int j = 0; j < 2; ++j) acc[i][j] = (v8f){0.f,0.f,0.f,0.f,0.f,0.f,0.f,0.f};

    const int r16 = lane & 15;
    const int hi  = lane >> 4;           // half-wave select
    const int kbA = hi << 3;             // A frag K interleave base: 0 or 8

    // per-thread staging chunks (8 bf16 = 16B each), 4 chunks cover 128x64
    int srow[4], scol[4];
#pragma unroll
    for (int c = 0; c < 4; ++c) {
        int idx = (c * 256 + tid) * 8;
        srow[c] = idx >> 6;
        scol[c] = idx & 63;
    }

    const int nk = K / BK;

    // prologue: fill buffer 0
#pragma unroll
    for (int c = 0; c < 4; ++c) {
        async_copy_b128(lds_addr(&Xs[0][srow[c] * LDSP + scol[c]]),
                        X + (size_t)(bm + srow[c]) * K + scol[c]);
        async_copy_b128(lds_addr(&Ws[0][srow[c] * LDSP + scol[c]]),
                        W + (size_t)(bn + srow[c]) * K + scol[c]);
    }

    for (int kt = 0; kt < nk; ++kt) {
        wait_async0();        // own async copies into buf[kt&1] have landed
        __syncthreads();      // everyone's copies visible; buf[(kt+1)&1] free

        if (kt + 1 < nk) {    // issue next tile while computing this one
            const int k0n = (kt + 1) * BK;
            const int nb  = (kt + 1) & 1;
#pragma unroll
            for (int c = 0; c < 4; ++c) {
                async_copy_b128(lds_addr(&Xs[nb][srow[c] * LDSP + scol[c]]),
                                X + (size_t)(bm + srow[c]) * K + k0n + scol[c]);
                async_copy_b128(lds_addr(&Ws[nb][srow[c] * LDSP + scol[c]]),
                                W + (size_t)(bn + srow[c]) * K + k0n + scol[c]);
            }
        }

        const __bf16* xs = Xs[kt & 1];
        const __bf16* ws = Ws[kt & 1];
#pragma unroll
        for (int ks = 0; ks < BK; ks += 32) {
            v16bf afr[4], bfr[2];
            // A fragments: lane=row(M), K split {kbA..+7, kbA+16..+23}
#pragma unroll
            for (int i = 0; i < 4; ++i) {
                int row = wm * 64 + i * 16 + r16;
                FragU f;
                f.u[0] = *(const uint4*)(&xs[row * LDSP + ks + kbA]);
                f.u[1] = *(const uint4*)(&xs[row * LDSP + ks + kbA + 16]);
                afr[i] = f.v;
            }
            // B fragments: lane=column(N = W row), K split {kb..+7, kb+16..+23} (matches A)
#pragma unroll
            for (int j = 0; j < 2; ++j) {
                int row = wn * 32 + j * 16 + r16;
                int ko  = ks + (hi << 3);
                FragU f;
                f.u[0] = *(const uint4*)(&ws[row * LDSP + ko]);
                f.u[1] = *(const uint4*)(&ws[row * LDSP + ko + 16]);
                bfr[j] = f.v;
            }
#pragma unroll
            for (int i = 0; i < 4; ++i)
#pragma unroll
                for (int j = 0; j < 2; ++j)
                    acc[i][j] = __builtin_amdgcn_wmma_f32_16x16x32_bf16(
                        false, afr[i], false, bfr[j],
                        (short)0, acc[i][j], false, false);
        }
        // no trailing barrier: next iteration's wait+barrier provides it
    }

    // ---- epilogue: C/D layout VGPR v -> M = v + 8*hi, N = lane&15 ----
    const int rowOff = hi << 3;
    const int col16  = lane & 15;
#pragma unroll
    for (int j = 0; j < 2; ++j) {
        int n = bn + wn * 32 + j * 16 + col16;
        float bv = bias[n];
#pragma unroll
        for (int i = 0; i < 4; ++i) {
            int mbase = bm + wm * 64 + i * 16 + rowOff;
#pragma unroll
            for (int v = 0; v < 8; ++v) {
                float o = acc[i][j][v] + bv;
                Y[(size_t)(mbase + v) * Nf + n] = (__bf16)o;
            }
        }
    }
}

// ------------------------------- launch ------------------------------------
extern "C" void kernel_launch(void* const* d_in, const int* in_sizes, int n_in,
                              void* d_out, int out_size, void* d_ws, size_t ws_size,
                              hipStream_t stream)
{
    (void)n_in; (void)ws_size;
    const __bf16* x     = (const __bf16*)d_in[0];
    const int*    pw    = (const int*)d_in[1];
    const float*  scale = (const float*)d_in[2];
    const float*  bias  = (const float*)d_in[3];
    // d_in[4] = K_orig (device scalar) -- derived from sizes instead:
    const int N       = in_sizes[3];            // 4096
    const int tokens  = out_size / N;           // 2048
    const int K       = in_sizes[0] / tokens;   // 4096
    const int K_packs = in_sizes[1] / N;        // 410

    __bf16* wd = (__bf16*)d_ws;                 // needs N*K*2 = 33.5 MB scratch

    const int total_words = N * K_packs;
    penta_dequant_kernel<<<(total_words + 255) / 256, 256, 0, stream>>>(
        pw, scale, wd, K, K_packs, total_words);

    dim3 grid(N / BN, tokens / BM);
    penta_gemm_kernel<<<grid, 256, 0, stream>>>(x, wd, bias, (__bf16*)d_out, N, K);
}
